// ModelDecoder_10582799417469
// MI455X (gfx1250) — matrix operand order repaired and checked by reading, not compiled, not run on hardware
//
#include <hip/hip_runtime.h>
#include <hip/hip_bf16.h>
#include <math.h>

// ---------------------------------------------------------------------------
// MI455X (gfx1250) implementation.
// Precision strategy: activations/weights in BF16, WMMA f32 accumulate
// (v_wmma_f32_16x16x32_bf16). FP32 matrix math would be compute-bound
// (~80 GFLOP vs ~45us/layer of HBM time at 23.3 TB/s); BF16 WMMA makes the
// whole pipeline memory-bound on the E x 64 edge-feature stream.
// Decoder big matmuls (x@xp_W, e@ep_W) are algebraically folded into 64x3 /
// 64x6 combined weights (associativity), removing ~38 GFLOP and ~1GB traffic.
// Edge/node WMMA kernels are persistent (grid-stride over 16-row wave-tiles)
// so the 24KB/16KB LDS weight panel is staged once per block, not once per
// 128 rows (saves ~190MB of redundant L2->LDS weight traffic).
// ---------------------------------------------------------------------------

typedef __bf16 bf16_t;
typedef __attribute__((ext_vector_type(16))) __bf16 v16bf;
typedef __attribute__((ext_vector_type(8)))  __bf16 v8bf;
typedef __attribute__((ext_vector_type(8)))  float  v8f;

#define G_  100
#define N_  100000
#define E_  1000000
#define H_  64
#define L_  3

union AV { v16bf v; v8bf h[2]; };

// Load one 16x32 bf16 A-operand chunk for the current lane, straight from a
// row-major 64-wide bf16 row in global memory, in the ISA-documented layout:
// lanes 0-15 hold row M=lane, K = kb+0..7 (v0..v3) and kb+16..23 (v4..v7);
// lanes 16-31 hold K = kb+8..15 and kb+24..31.  -> two 16-byte loads.
__device__ __forceinline__ v16bf load_a_chunk(const bf16_t* __restrict__ row,
                                              int kb, int lane) {
  int koff = kb + ((lane & 16) ? 8 : 0);
  AV u;
  u.h[0] = *(const v8bf*)(row + koff);
  u.h[1] = *(const v8bf*)(row + koff + 16);
  return u.v;
}

// ---------------------------------------------------------------------------
// Tiny per-graph MLP: g = relu(ge@W1+b1)@W2+b2 ; split into node/edge feats.
__global__ __launch_bounds__(64) void k_gmlp(
    const float* __restrict__ ge, const float* __restrict__ W1,
    const float* __restrict__ b1, const float* __restrict__ W2,
    const float* __restrict__ b2, float* __restrict__ nodef,
    float* __restrict__ edgef) {
  __shared__ float gin[H_], h[H_];
  int g = blockIdx.x, t = threadIdx.x;
  gin[t] = ge[g * H_ + t];
  __syncthreads();
  float a = b1[t];
  for (int k = 0; k < H_; ++k) a += gin[k] * W1[k * H_ + t];
  h[t] = fmaxf(a, 0.f);
  __syncthreads();
  float o0 = b2[t], o1 = b2[H_ + t];
  for (int k = 0; k < H_; ++k) {
    o0 += h[k] * W2[k * 2 * H_ + t];
    o1 += h[k] * W2[k * 2 * H_ + H_ + t];
  }
  nodef[g * H_ + t] = o0;
  edgef[g * H_ + t] = o1;
}

// ---------------------------------------------------------------------------
// Repack We (L,192,64) and Wn (L,128,64) f32 -> bf16 in WMMA B-operand layout:
// [layer][kchunk][ntile][lane][i] with lane's 16 values = K rows
// kb + (lane>=16?8:0) + (i<8 ? i : i+8) at column n*16 + lane%16.
__global__ void k_wprep(const float* __restrict__ We, const float* __restrict__ Wn,
                        bf16_t* __restrict__ Web, bf16_t* __restrict__ Wnb) {
  const int tot_e = L_ * 6 * 4 * 512;   // 36864
  const int tot_n = L_ * 4 * 4 * 512;   // 24576
  for (int t = blockIdx.x * blockDim.x + threadIdx.x; t < tot_e + tot_n;
       t += gridDim.x * blockDim.x) {
    if (t < tot_e) {
      int i = t & 15, ln = (t >> 4) & 31, n = (t >> 9) & 3;
      int k = (t >> 11) % 6, l = t / (6 * 4 * 512);
      int K = k * 32 + ((ln & 16) ? 8 : 0) + (i < 8 ? i : i + 8);
      int col = n * 16 + (ln & 15);
      Web[t] = (bf16_t)We[(l * 192 + K) * 64 + col];
    } else {
      int u = t - tot_e;
      int i = u & 15, ln = (u >> 4) & 31, n = (u >> 9) & 3;
      int k = (u >> 11) & 3, l = u / (4 * 4 * 512);
      int K = k * 32 + ((ln & 16) ? 8 : 0) + (i < 8 ? i : i + 8);
      int col = n * 16 + (ln & 15);
      Wnb[u] = (bf16_t)Wn[(l * 128 + K) * 64 + col];
    }
  }
}

// ---------------------------------------------------------------------------
// Fold decoder: Wnode[k][c] = sum_j xp_W[k, c*64+j] * vec_c[j]  (c=0..2)
//               Wedge[k][c] similarly from ep_W (c=0..2) and cp_W (c=3..5).
__global__ void k_fold(
    const float* __restrict__ xp_W, const float* __restrict__ xp_b,
    const float* __restrict__ nl_W, const float* __restrict__ nl_b,
    const float* __restrict__ nn_W, const float* __restrict__ nn_b,
    const float* __restrict__ nt_W, const float* __restrict__ nt_b,
    const float* __restrict__ ep_W, const float* __restrict__ ep_b,
    const float* __restrict__ we_W, const float* __restrict__ we_b,
    const float* __restrict__ el_W, const float* __restrict__ el_b,
    const float* __restrict__ et_W, const float* __restrict__ et_b,
    const float* __restrict__ cp_W, const float* __restrict__ cp_b,
    float* __restrict__ Wnd, float* __restrict__ bnd,
    float* __restrict__ Wed, float* __restrict__ bed) {
  int t = threadIdx.x;
  const float* nvec[3] = {nl_W, nn_W, nt_W};
  const float* nbia[3] = {nl_b, nn_b, nt_b};
  const float* evec[3] = {we_W, el_W, et_W};
  const float* ebia[3] = {we_b, el_b, et_b};
  if (t < 64 * 3) {                         // node weights
    int k = t / 3, c = t % 3;
    float a = 0.f;
    for (int j = 0; j < 64; ++j) a += xp_W[k * 192 + c * 64 + j] * nvec[c][j];
    Wnd[k * 3 + c] = a;
  } else if (t < 64 * 3 + 64 * 6) {         // edge weights
    int u = t - 64 * 3;
    int k = u / 6, c = u % 6;
    float a = 0.f;
    if (c < 3) {
      for (int j = 0; j < 64; ++j) a += ep_W[k * 256 + c * 64 + j] * evec[c][j];
    } else {
      int cc = c - 3;
      for (int j = 0; j < 64; ++j) a += ep_W[k * 256 + 192 + j] * cp_W[j * 3 + cc];
    }
    Wed[k * 6 + c] = a;
  } else if (t < 64 * 9 + 3) {              // node biases
    int c = t - 64 * 9;
    float a = nbia[c][0];
    for (int j = 0; j < 64; ++j) a += xp_b[c * 64 + j] * nvec[c][j];
    bnd[c] = a;
  } else if (t < 64 * 9 + 9) {              // edge biases
    int c = t - 64 * 9 - 3;
    float a;
    if (c < 3) {
      a = ebia[c][0];
      for (int j = 0; j < 64; ++j) a += ep_b[c * 64 + j] * evec[c][j];
    } else {
      int cc = c - 3;
      a = cp_b[cc];
      for (int j = 0; j < 64; ++j) a += ep_b[192 + j] * cp_W[j * 3 + cc];
    }
    bed[c] = a;
  }
}

// ---------------------------------------------------------------------------
// Unpool gathers (node table is L2-resident: 25.6 KB << 192 MB L2).
__global__ void k_unpool_x(const float* __restrict__ nodef,
                           const int* __restrict__ nb,
                           bf16_t* __restrict__ x, long total) {
  for (long t = blockIdx.x * (long)blockDim.x + threadIdx.x; t < total;
       t += (long)gridDim.x * blockDim.x) {
    long i = t >> 6;
    int c = (int)(t & 63);
    x[t] = (bf16_t)nodef[(long)nb[i] * 64 + c];
  }
}

__global__ void k_unpool_e(const float* __restrict__ edgef,
                           const int* __restrict__ nb,
                           const int* __restrict__ src,
                           bf16_t* __restrict__ e, long total) {
  for (long t = blockIdx.x * (long)blockDim.x + threadIdx.x; t < total;
       t += (long)gridDim.x * blockDim.x) {
    long j = t >> 6;
    int c = (int)(t & 63);
    e[t] = (bf16_t)edgef[(long)nb[src[j]] * 64 + c];
  }
}

__global__ void k_zero(float* __restrict__ p, long total) {
  for (long t = blockIdx.x * (long)blockDim.x + threadIdx.x; t < total;
       t += (long)gridDim.x * blockDim.x)
    p[t] = 0.f;
}

// ---------------------------------------------------------------------------
// Edge MLP layer: msg = relu([x_src|x_dst|e] @ We + be); agg += scatter(msg);
// e <- msg (bf16, in place: each 16-edge tile reads then rewrites its rows).
// Persistent: each wave grid-strides over 16-edge tiles; A gathered directly
// from global in WMMA layout; B (192x64 bf16) staged ONCE per block in LDS.
__global__ __launch_bounds__(256) void k_edge(
    const bf16_t* __restrict__ x, bf16_t* __restrict__ e,
    float* __restrict__ agg, const bf16_t* __restrict__ Wb,
    const float* __restrict__ be, const int* __restrict__ src,
    const int* __restrict__ dst, int nedge) {
  __shared__ bf16_t Blds[6 * 4 * 512];        // 24 KB
  __shared__ float  stage[8][16 * 64];        // 32 KB (wave-private slices)
  for (int i = threadIdx.x; i < 6 * 4 * 512 / 8; i += 256)
    ((v8bf*)Blds)[i] = ((const v8bf*)Wb)[i];
  __syncthreads();                            // only barrier in the kernel

  const int wave = threadIdx.x >> 5, lane = threadIdx.x & 31;
  const int m = lane & 15;
  const int colb = lane & 15;
  const int rbase = (lane & 16) ? 8 : 0;
  const int srow = lane >> 1, scb = (lane & 1) * 32;   // writeback mapping
  float* st = stage[wave];

  // per-lane bias values are loop-invariant
  float bias[4];
#pragma unroll
  for (int n = 0; n < 4; ++n) bias[n] = be[n * 16 + colb];

  const int ntiles = nedge >> 4;              // E/16 (exact: E % 16 == 0)
  const int tstride = gridDim.x * 8;
  for (int tile = blockIdx.x * 8 + wave; tile < ntiles; tile += tstride) {
    long base = (long)tile * 16;
    long j = base + m;
    int sj = src[j], dj = dst[j];
    const bf16_t* rx_s = x + (long)sj * 64;
    const bf16_t* rx_d = x + (long)dj * 64;
    const bf16_t* re   = e + j * 64;
    // prefetch this wave's NEXT tile of e rows (stream stride)
    long jn = j + (long)tstride * 16;
    if (jn < nedge) __builtin_prefetch(e + jn * 64, 0, 1);

    v8f c[4] = {};
#pragma unroll
    for (int k = 0; k < 6; ++k) {
      const bf16_t* row = (k < 2) ? rx_s : (k < 4) ? rx_d : re;
      v16bf a = load_a_chunk(row, (k & 1) * 32, lane);
      const bf16_t* bp = Blds + k * 4 * 512;
#pragma unroll
      for (int n = 0; n < 4; ++n) {
        v16bf b = *(const v16bf*)(bp + n * 512 + lane * 16);
        c[n] = __builtin_amdgcn_wmma_f32_16x16x32_bf16(
            false, a, false, b, (short)0, c[n], false, false);
      }
    }

    // bias + relu; scatter-add into agg (L2 atomics); stage f32 tile in LDS.
#pragma unroll
    for (int n = 0; n < 4; ++n) {
      int col = n * 16 + colb;
#pragma unroll
      for (int r = 0; r < 8; ++r) {
        float v = c[n][r] + bias[n];
        v = v > 0.f ? v : 0.f;
        int row = rbase + r;
        st[row * 64 + col] = v;
        int d = __shfl(dj, row, 32);          // dst of that tile row
        atomicAdd(agg + (long)d * 64 + col, v);
      }
    }
    // packed bf16 writeback of msg -> e (row-major), 64 B per lane.
    bf16_t* erow = e + (base + srow) * 64 + scb;
#pragma unroll
    for (int q = 0; q < 32; q += 8) {
      v8bf o;
#pragma unroll
      for (int t = 0; t < 8; ++t) o[t] = (bf16_t)st[srow * 64 + scb + q + t];
      *(v8bf*)(erow + q) = o;
    }
  }
}

// ---------------------------------------------------------------------------
// Node MLP layer: x_next = relu([x|agg] @ Wn + bn). Persistent, one wave per
// 16-node tile per iteration.
__global__ __launch_bounds__(256) void k_node(
    const bf16_t* __restrict__ x, const float* __restrict__ agg,
    bf16_t* __restrict__ xn, const bf16_t* __restrict__ Wb,
    const float* __restrict__ bn, int nnode) {
  __shared__ bf16_t Blds[4 * 4 * 512];        // 16 KB
  __shared__ float  stage[8][16 * 64];        // 32 KB
  for (int i = threadIdx.x; i < 4 * 4 * 512 / 8; i += 256)
    ((v8bf*)Blds)[i] = ((const v8bf*)Wb)[i];
  __syncthreads();

  const int wave = threadIdx.x >> 5, lane = threadIdx.x & 31;
  const int m = lane & 15;
  const int colb = lane & 15;
  const int rbase = (lane & 16) ? 8 : 0;
  const int srow = lane >> 1, scb = (lane & 1) * 32;
  float* st = stage[wave];

  float bias[4];
#pragma unroll
  for (int n = 0; n < 4; ++n) bias[n] = bn[n * 16 + colb];

  const int ntiles = nnode >> 4;              // N/16 (exact: N % 16 == 0)
  const int tstride = gridDim.x * 8;
  for (int tile = blockIdx.x * 8 + wave; tile < ntiles; tile += tstride) {
    long base = (long)tile * 16;
    long i = base + m;
    const bf16_t* xr = x + i * 64;
    const float*  ar = agg + i * 64;

    v8f c[4] = {};
#pragma unroll
    for (int k = 0; k < 4; ++k) {
      v16bf a;
      if (k < 2) {
        a = load_a_chunk(xr, k * 32, lane);
      } else {                                // f32 agg -> bf16 on the fly
        int koff = (k - 2) * 32 + ((lane & 16) ? 8 : 0);
        AV u;
#pragma unroll
        for (int t = 0; t < 8; ++t) {
          u.h[0][t] = (bf16_t)ar[koff + t];
          u.h[1][t] = (bf16_t)ar[koff + 16 + t];
        }
        a = u.v;
      }
      const bf16_t* bp = Blds + k * 4 * 512;
#pragma unroll
      for (int n = 0; n < 4; ++n) {
        v16bf b = *(const v16bf*)(bp + n * 512 + lane * 16);
        c[n] = __builtin_amdgcn_wmma_f32_16x16x32_bf16(
            false, a, false, b, (short)0, c[n], false, false);
      }
    }

#pragma unroll
    for (int n = 0; n < 4; ++n) {
      int col = n * 16 + colb;
#pragma unroll
      for (int r = 0; r < 8; ++r) {
        float v = c[n][r] + bias[n];
        st[(rbase + r) * 64 + col] = v > 0.f ? v : 0.f;
      }
    }
    bf16_t* orow = xn + (base + srow) * 64 + scb;
#pragma unroll
    for (int q = 0; q < 32; q += 8) {
      v8bf o;
#pragma unroll
      for (int t = 0; t < 8; ++t) o[t] = (bf16_t)st[srow * 64 + scb + q + t];
      *(v8bf*)(orow + q) = o;
    }
  }
}

// ---------------------------------------------------------------------------
// Folded decoder: out[r][c] = maybe_sin( feat[r,:] . Wc[:,c] + bc[c] ).
__global__ __launch_bounds__(256) void k_dec(
    const bf16_t* __restrict__ feat, const float* __restrict__ Wc,
    const float* __restrict__ bc, float* __restrict__ out, long R, int C,
    unsigned sinmask) {
  __shared__ float Wl[64 * 6];
  __shared__ float bl[8];
  for (int i = threadIdx.x; i < 64 * C; i += blockDim.x) Wl[i] = Wc[i];
  if (threadIdx.x < (unsigned)C) bl[threadIdx.x] = bc[threadIdx.x];
  __syncthreads();
  long r = blockIdx.x * (long)blockDim.x + threadIdx.x;
  if (r >= R) return;
  const v8bf* row = (const v8bf*)(feat + r * 64);
  float acc[6];
  for (int cidx = 0; cidx < C; ++cidx) acc[cidx] = bl[cidx];
#pragma unroll
  for (int g = 0; g < 8; ++g) {
    v8bf s = row[g];
#pragma unroll
    for (int t = 0; t < 8; ++t) {
      float f = (float)s[t];
      int k = g * 8 + t;
      for (int cidx = 0; cidx < C; ++cidx) acc[cidx] += f * Wl[k * C + cidx];
    }
  }
  for (int cidx = 0; cidx < C; ++cidx) {
    float v = acc[cidx];
    if ((sinmask >> cidx) & 1u) v = sinf(v);
    out[r * C + cidx] = v;
  }
}

// ---------------------------------------------------------------------------
extern "C" void kernel_launch(void* const* d_in, const int* in_sizes, int n_in,
                              void* d_out, int out_size, void* d_ws,
                              size_t ws_size, hipStream_t stream) {
  const float* ge     = (const float*)d_in[0];
  const int*   nbatch = (const int*)d_in[1];
  const int*   eidx   = (const int*)d_in[2];
  const float* W1 = (const float*)d_in[3];  const float* b1 = (const float*)d_in[4];
  const float* W2 = (const float*)d_in[5];  const float* b2 = (const float*)d_in[6];
  const float* We = (const float*)d_in[7];  const float* be = (const float*)d_in[8];
  const float* Wn = (const float*)d_in[9];  const float* bn = (const float*)d_in[10];
  const float* xp_W = (const float*)d_in[11]; const float* xp_b = (const float*)d_in[12];
  const float* nl_W = (const float*)d_in[13]; const float* nl_b = (const float*)d_in[14];
  const float* nn_W = (const float*)d_in[15]; const float* nn_b = (const float*)d_in[16];
  const float* nt_W = (const float*)d_in[17]; const float* nt_b = (const float*)d_in[18];
  const float* ep_W = (const float*)d_in[19]; const float* ep_b = (const float*)d_in[20];
  const float* we_W = (const float*)d_in[21]; const float* we_b = (const float*)d_in[22];
  const float* el_W = (const float*)d_in[23]; const float* el_b = (const float*)d_in[24];
  const float* et_W = (const float*)d_in[25]; const float* et_b = (const float*)d_in[26];
  const float* cp_W = (const float*)d_in[27]; const float* cp_b = (const float*)d_in[28];

  const int* src = eidx;        // edge_index[0]
  const int* dst = eidx + E_;   // edge_index[1]

  // --- carve workspace -----------------------------------------------------
  char* w = (char*)d_ws;
  size_t off = 0;
  auto take = [&](size_t bytes) -> void* {
    void* p = w + off;
    off = (off + bytes + 255) & ~(size_t)255;
    return p;
  };
  float*  nodef = (float*)take(G_ * H_ * sizeof(float));
  float*  edgef = (float*)take(G_ * H_ * sizeof(float));
  bf16_t* Web   = (bf16_t*)take(L_ * 6 * 4 * 512 * sizeof(bf16_t));
  bf16_t* Wnb   = (bf16_t*)take(L_ * 4 * 4 * 512 * sizeof(bf16_t));
  float*  Wnd   = (float*)take(64 * 3 * sizeof(float));
  float*  bnd   = (float*)take(8 * sizeof(float));
  float*  Wed   = (float*)take(64 * 6 * sizeof(float));
  float*  bed   = (float*)take(8 * sizeof(float));
  bf16_t* x0    = (bf16_t*)take((size_t)N_ * H_ * sizeof(bf16_t));
  bf16_t* x1    = (bf16_t*)take((size_t)N_ * H_ * sizeof(bf16_t));
  bf16_t* ebf   = (bf16_t*)take((size_t)E_ * H_ * sizeof(bf16_t));
  float*  agg   = (float*)take((size_t)N_ * H_ * sizeof(float));
  (void)ws_size; (void)in_sizes; (void)n_in; (void)out_size;

  // --- prep ----------------------------------------------------------------
  k_gmlp<<<G_, 64, 0, stream>>>(ge, W1, b1, W2, b2, nodef, edgef);
  k_wprep<<<96, 256, 0, stream>>>(We, Wn, Web, Wnb);
  k_fold<<<1, 640, 0, stream>>>(xp_W, xp_b, nl_W, nl_b, nn_W, nn_b, nt_W, nt_b,
                                ep_W, ep_b, we_W, we_b, el_W, el_b, et_W, et_b,
                                cp_W, cp_b, Wnd, bnd, Wed, bed);
  {
    long tx = (long)N_ * H_;
    k_unpool_x<<<(int)((tx + 255) / 256), 256, 0, stream>>>(nodef, nbatch, x0, tx);
    long te = (long)E_ * H_;
    k_unpool_e<<<(int)((te + 255) / 256), 256, 0, stream>>>(edgef, nbatch, src, ebf, te);
  }

  // --- 3 message-passing layers (persistent WMMA kernels) ------------------
  bf16_t* xc = x0;
  bf16_t* xn = x1;
  const int eblocks = 2048;   // 16K waves grid-striding 62500 tiles (~4/wave)
  const int nblocks = 512;    // 4K waves grid-striding 6250 tiles
  for (int l = 0; l < L_; ++l) {
    long ta = (long)N_ * H_;
    k_zero<<<(int)((ta + 255) / 256), 256, 0, stream>>>(agg, ta);
    k_edge<<<eblocks, 256, 0, stream>>>(xc, ebf, agg, Web + l * 6 * 4 * 512,
                                        be + l * H_, src, dst, E_);
    k_node<<<nblocks, 256, 0, stream>>>(xc, agg, xn, Wnb + l * 4 * 4 * 512,
                                        bn + l * H_, N_);
    bf16_t* t = xc; xc = xn; xn = t;
  }

  // --- folded decoders -----------------------------------------------------
  float* out = (float*)d_out;
  k_dec<<<(N_ + 255) / 256, 256, 0, stream>>>(xc, Wnd, bnd, out, N_, 3, 0x3u);
  k_dec<<<(E_ + 255) / 256, 256, 0, stream>>>(ebf, Wed, bed, out + (long)N_ * 3,
                                              E_, 6, 0x3Bu);
}
